// BaselineModel_53274774340238
// MI455X (gfx1250) — matrix operand order repaired and checked by reading, not compiled, not run on hardware
//
#include <hip/hip_runtime.h>
#include <hip/hip_bf16.h>
#include <stdint.h>

// ---------------------------------------------------------------------------
// DeepFM inference for MI455X (gfx1250, wave32).
//   B=16384 batch, F=26 fields, V=100000 vocab, D=64, H1=256, H2=128.
// Roofline: gather = 109 MB @ 23.3 TB/s ~ 4.7us (dominant, irreducible).
//           MLP = ~1.6 GFLOP -> f16 WMMA (v_wmma_f32_16x16x32_f16), trivial.
// Sigmoid uses v_exp_f32 + v_rcp_f32 directly (no precise-divide chain).
// ---------------------------------------------------------------------------

typedef _Float16 f16;
typedef __attribute__((ext_vector_type(16))) _Float16 v16h;
typedef __attribute__((ext_vector_type(8)))  _Float16 v8h;
typedef __attribute__((ext_vector_type(8)))  float    v8f;

#define B_  16384
#define F_  26
#define V_  100000
#define D_  64
#define H1_ 256
#define H2_ 128

union V16 { v16h v; v8h h[2]; };

// sigmoid(x) = rcp(1 + exp2(-x*log2(e)))  -> v_mul, v_exp_f32, v_add, v_rcp_f32
__device__ __forceinline__ float sigmoidf_fast(float x) {
  float e = __builtin_amdgcn_exp2f(x * -1.44269504088896340736f);
  return __builtin_amdgcn_rcpf(1.0f + e);
}

// ---------------------------------------------------------------------------
// Kernel 0: transpose weights to column-major f16 so WMMA B-fragments are
// contiguous 32B per lane.  W1t[n*64+k], W2t[n*256+k].
// ---------------------------------------------------------------------------
__global__ __launch_bounds__(256) void prep_weights(
    const float* __restrict__ Wh1, const float* __restrict__ Wh2,
    f16* __restrict__ W1t, f16* __restrict__ W2t) {
  int i = blockIdx.x * 256 + threadIdx.x;
  if (i < H1_ * D_) {                 // 16384: n = i/64, k = i%64
    int n = i >> 6, k = i & 63;
    W1t[i] = (f16)Wh1[k * H1_ + n];
  }
  if (i < H2_ * H1_) {                // 32768: n = i/256, k = i%256
    int n = i >> 8, k = i & 255;
    W2t[i] = (f16)Wh2[k * H2_ + n];
  }
}

// ---------------------------------------------------------------------------
// Kernel 1: fused embedding gather + FM second-order.  One wave per batch
// row; lane handles dims {2l, 2l+1} via float2 -> fully coalesced 256B rows.
// Emits fm (f16, MLP input) and fm row-sum (f32, needed for final output).
// ---------------------------------------------------------------------------
__global__ __launch_bounds__(256) void gather_fm(
    const int* __restrict__ cat, const float* __restrict__ W2,
    f16* __restrict__ fm16, float* __restrict__ fmsum) {
  const int lane = threadIdx.x & 31;
  const int wave = threadIdx.x >> 5;
  const int row  = blockIdx.x * 8 + wave;

  // one load of all 26 indices across lanes, broadcast via shuffle
  const int* crow = cat + (size_t)row * F_;
  int myidx = (lane < F_) ? crow[lane] : 0;

  float sx = 0.f, sy = 0.f, qx = 0.f, qy = 0.f;
  #pragma unroll
  for (int f = 0; f < F_; ++f) {
    int idx = __shfl(myidx, f, 32);
    const float2 e2 = *(const float2*)(W2 + ((size_t)f * V_ + (size_t)idx) * D_ + lane * 2);
    float ex = e2.x * 0.1f, ey = e2.y * 0.1f;
    sx += ex; sy += ey; qx += ex * ex; qy += ey * ey;
  }
  float f0 = 0.5f * (sx * sx - qx);
  float f1 = 0.5f * (sy * sy - qy);

  union { f16 h[2]; uint32_t u; } p;
  p.h[0] = (f16)f0; p.h[1] = (f16)f1;
  ((uint32_t*)fm16)[(size_t)row * (D_ / 2) + lane] = p.u;

  float r = f0 + f1;
  #pragma unroll
  for (int m = 16; m >= 1; m >>= 1) r += __shfl_xor(r, m, 32);
  if (lane == 0) fmsum[row] = r;
}

// ---------------------------------------------------------------------------
// Kernel 2: fused sigmoid-MLP via WMMA.  Each wave owns 16 batch rows.
//   h1 = sig(fm@W1+b1)  : 16 N-tiles x 2 k-steps  = 32 WMMA
//   h2 = sig(h1@W2+b2)  :  8 N-tiles x 8 k-steps  = 64 WMMA
//   out = h2@Wout + bout + fmsum + bias : per-lane FMA + 16-lane shfl reduce
// h1 is staged through a wave-private LDS slice (C-layout -> A-layout
// transpose); no block barriers required.
// ---------------------------------------------------------------------------
__global__ __launch_bounds__(256) void mlp_wmma(
    const f16* __restrict__ fm16, const float* __restrict__ fmsum,
    const f16* __restrict__ W1t, const float* __restrict__ b1,
    const f16* __restrict__ W2t, const float* __restrict__ b2,
    const float* __restrict__ Wout, const float* __restrict__ bout,
    const float* __restrict__ bias, float* __restrict__ out) {
  __shared__ f16 lds_h1[8][16][H1_];          // 64 KB, per-wave 8 KB slices

  const int lane = threadIdx.x & 31;
  const int wave = threadIdx.x >> 5;
  const int m0   = blockIdx.x * 128 + wave * 16;
  const int lm   = lane & 15;                 // A: row M / B,C: col N
  const int lg   = lane >> 4;                 // lane half (K-group select)

  // ---- A fragments of fm (16x64): this code's K order (matches B):
  // elems 0-15 = K kb..kb+15 (contiguous, kb=16*lg)
  const f16* arow = fm16 + (size_t)(m0 + lm) * D_ + lg * 16;
  V16 a0, a1;
  a0.h[0] = *(const v8h*)(arow +  0);
  a0.h[1] = *(const v8h*)(arow +  8);
  a1.h[0] = *(const v8h*)(arow + 32);
  a1.h[1] = *(const v8h*)(arow + 40);

  // ---- GEMM1 + sigmoid -> LDS (row-major [16][256] per wave)
  #pragma unroll 2
  for (int t = 0; t < 16; ++t) {
    const int n = t * 16 + lm;
    // B layout: N = lm per lane; elems 0-15 = K lg*16 .. lg*16+15 (contig)
    const f16* bp = W1t + (size_t)n * D_ + lg * 16;
    v16h b0 = *(const v16h*)(bp);
    v16h bK = *(const v16h*)(bp + 32);
    v8f c = {};
    c = __builtin_amdgcn_wmma_f32_16x16x32_f16(false, a0.v, false, b0, (short)0, c, false, false);
    c = __builtin_amdgcn_wmma_f32_16x16x32_f16(false, a1.v, false, bK, (short)0, c, false, false);
    const float bb = b1[n];
    #pragma unroll
    for (int r = 0; r < 8; ++r) {
      // C layout: VGPR r -> M = r + 8*lg, lane -> N
      lds_h1[wave][r + 8 * lg][n] = (f16)sigmoidf_fast(c[r] + bb);
    }
  }

  // ---- GEMM2 + sigmoid, fused with @Wout column reduction
  float accM[8] = {0.f, 0.f, 0.f, 0.f, 0.f, 0.f, 0.f, 0.f};
  const f16* h1p = &lds_h1[wave][0][0];
  #pragma unroll 2
  for (int t = 0; t < 8; ++t) {
    const int n = t * 16 + lm;
    const f16* bbase = W2t + (size_t)n * H1_;
    v8f c2 = {};
    #pragma unroll
    for (int kk = 0; kk < 8; ++kk) {
      V16 aa;
      const f16* ap = h1p + lm * H1_ + kk * 32 + lg * 16;
      aa.h[0] = *(const v8h*)(ap);
      aa.h[1] = *(const v8h*)(ap + 8);
      v16h bw = *(const v16h*)(bbase + kk * 32 + lg * 16);
      c2 = __builtin_amdgcn_wmma_f32_16x16x32_f16(false, aa.v, false, bw, (short)0, c2, false, false);
    }
    const float bb2 = b2[n];
    const float wn  = Wout[n];
    #pragma unroll
    for (int r = 0; r < 8; ++r)
      accM[r] += sigmoidf_fast(c2[r] + bb2) * wn;   // partial over this lane's N
  }

  // reduce over the 16 lanes sharing each M (lanes 0-15 -> M=r, 16-31 -> r+8)
  const float cb = bout[0] + bias[0];
  #pragma unroll
  for (int r = 0; r < 8; ++r) {
    float a = accM[r];
    a += __shfl_xor(a, 8, 16);
    a += __shfl_xor(a, 4, 16);
    a += __shfl_xor(a, 2, 16);
    a += __shfl_xor(a, 1, 16);
    if (lm == 0) {
      const int row = m0 + r + 8 * lg;
      out[row] = a + cb + fmsum[row];
    }
  }
}

// ---------------------------------------------------------------------------
extern "C" void kernel_launch(void* const* d_in, const int* in_sizes, int n_in,
                              void* d_out, int out_size, void* d_ws, size_t ws_size,
                              hipStream_t stream) {
  const int*   cat  = (const int*)  d_in[0];   // cat_feat [B,F]
  const float* W2   = (const float*)d_in[1];   // [F,V,D]
  const float* Wh1  = (const float*)d_in[2];   // [D,H1]
  const float* bh1  = (const float*)d_in[3];   // [H1]
  const float* Wh2  = (const float*)d_in[4];   // [H1,H2]
  const float* bh2  = (const float*)d_in[5];   // [H2]
  const float* Wout = (const float*)d_in[6];   // [H2,1]
  const float* bo   = (const float*)d_in[7];   // [1]
  const float* bias = (const float*)d_in[8];   // [1]
  float* out = (float*)d_out;                  // [B,1]

  char* ws = (char*)d_ws;
  f16*   fm16  = (f16*)ws;                                    // 2 MiB
  float* fmsum = (float*)(ws + (size_t)B_ * D_ * 2);          // 64 KiB
  f16*   W1t   = (f16*)(ws + (size_t)B_ * D_ * 2 + B_ * 4);   // 32 KiB
  f16*   W2t   = W1t + H1_ * D_;                              // 64 KiB

  prep_weights<<<128, 256, 0, stream>>>(Wh1, Wh2, W1t, W2t);
  gather_fm<<<B_ / 8, 256, 0, stream>>>(cat, W2, fm16, fmsum);
  mlp_wmma<<<B_ / 128, 256, 0, stream>>>(fm16, fmsum, W1t, bh1, W2t, bh2,
                                         Wout, bo, bias, out);
}
